// FSAGE_15358803051093
// MI455X (gfx1250) — matrix operand order repaired and checked by reading, not compiled, not run on hardware
//
#include <hip/hip_runtime.h>

typedef __attribute__((ext_vector_type(16))) __bf16 v16bf;
typedef __attribute__((ext_vector_type(8)))  __bf16 v8bf;
typedef __attribute__((ext_vector_type(8)))  float  v8f;

#define DIM    256
#define NNEI   10
#define MT     32         // nodes (rows) per block = two 16-row WMMA tiles
#define LDA    520        // padded LDS row stride in bf16 elements (512 + 8)
#define NNODES 100000

union ABfrag { v16bf v; v8bf h[2]; };

// ---- 32-contiguous-element row loaders (f32 or bf16 source) -> float[32] ----
__device__ __forceinline__ void load32(const float* __restrict__ p, float* v) {
  const float4* q = (const float4*)p;
#pragma unroll
  for (int i = 0; i < 8; ++i) {
    float4 t = q[i];
    v[4*i+0]=t.x; v[4*i+1]=t.y; v[4*i+2]=t.z; v[4*i+3]=t.w;
  }
}
__device__ __forceinline__ void load32(const __bf16* __restrict__ p, float* v) {
  const v8bf* q = (const v8bf*)p;
#pragma unroll
  for (int i = 0; i < 4; ++i) {
    v8bf t = q[i];
#pragma unroll
    for (int e = 0; e < 8; ++e) v[8*i+e] = (float)t[e];
  }
}

// ---- pack W [512,256] f32 (row-major, x@W) into per-lane bf16 B-fragments ----
// Fragment (kt, nt): 32 lanes x 16 bf16. lane l, element e holds
//   W[ kt*32 + (l>>4)*16 + e ][ nt*16 + (l&15) ]
__global__ void pack_w_kernel(const float* __restrict__ W, __bf16* __restrict__ Wp) {
  int t    = blockIdx.x * 256 + threadIdx.x;   // 0 .. 131071
  int e    = t & 15;
  int lane = (t >> 4) & 31;
  int kt   = (t >> 9) & 15;
  int nt   = t >> 13;
  int col  = nt * 16 + (lane & 15);
  int k    = kt * 32 + ((lane >> 4) << 4) + e;
  Wp[t] = (__bf16)W[(size_t)k * DIM + col];
}

// ---- fused gather + mean + concat + GEMM(+bias,+ReLU) for one layer ----
template <typename TIn, typename TOut, bool RELU>
__global__ __launch_bounds__(128) void sage_layer_kernel(
    const TIn*  __restrict__ h,
    const int*  __restrict__ adj,
    const __bf16* __restrict__ Wp,
    const float* __restrict__ bias,
    TOut* __restrict__ out)
{
  __shared__ __bf16 sA[MT * LDA];

  const int  tid = threadIdx.x;
  const long g0  = (long)blockIdx.x * MT;

  // ---------- Phase 1: build A = [self | neigh-mean] (32 x 512 bf16) in LDS ----------
  // 8 threads per node row, two passes of 16 rows.
#pragma unroll
  for (int half = 0; half < 2; ++half) {
    const int  node = (tid >> 3) + half * 16;   // 0..31
    const int  cg   = tid & 7;                  // 32-col group
    const long g    = g0 + node;
    float buf[32];

    // self embedding -> sA[node][0..255]
    load32(h + (size_t)g * DIM + cg * 32, buf);
    {
      __bf16* dst = &sA[node * LDA + cg * 32];
#pragma unroll
      for (int i = 0; i < 4; ++i) {
        v8bf t;
#pragma unroll
        for (int e = 0; e < 8; ++e) t[e] = (__bf16)buf[8*i+e];
        *(v8bf*)(dst + 8*i) = t;
      }
    }

    // neighbor mean -> sA[node][256..511]
    float acc[32];
#pragma unroll
    for (int i = 0; i < 32; ++i) acc[i] = 0.f;
    for (int n = 0; n < NNEI; ++n) {
      int idx = adj[g * NNEI + n];
      load32(h + (size_t)idx * DIM + cg * 32, buf);
#pragma unroll
      for (int i = 0; i < 32; ++i) acc[i] += buf[i];
    }
    {
      __bf16* dst = &sA[node * LDA + DIM + cg * 32];
#pragma unroll
      for (int i = 0; i < 4; ++i) {
        v8bf t;
#pragma unroll
        for (int e = 0; e < 8; ++e) t[e] = (__bf16)(acc[8*i+e] * 0.1f);
        *(v8bf*)(dst + 8*i) = t;
      }
    }
  }
  __syncthreads();

  // ---------- Phase 2: [32 x 512] @ [512 x 256] via v_wmma_f32_16x16x32_bf16 ----------
  // Each wave: 4 N-tiles x 2 M-tiles; every B fragment feeds two WMMAs.
  const int wave  = tid >> 5;              // 0..3 -> N-tiles wave*4 .. wave*4+3
  const int lane  = tid & 31;
  const int mrow  = lane & 15;             // A row within a 16-row tile
  const int khalf = (lane >> 4) << 4;      // 0 or 16 (A lane K-phase)

  v8f C[2][4];
#pragma unroll
  for (int m = 0; m < 2; ++m)
#pragma unroll
    for (int j = 0; j < 4; ++j)
#pragma unroll
      for (int r = 0; r < 8; ++r) C[m][j][r] = 0.f;

  for (int kt = 0; kt < 16; ++kt) {
    ABfrag A0, A1;
    const __bf16* ap0 = &sA[mrow * LDA + kt * 32 + khalf];
    const __bf16* ap1 = ap0 + 16 * LDA;
    A0.h[0] = *(const v8bf*)(ap0);          // K = khalf .. khalf+7
    A0.h[1] = *(const v8bf*)(ap0 + 8);      // K = khalf+8 .. khalf+15
    A1.h[0] = *(const v8bf*)(ap1);
    A1.h[1] = *(const v8bf*)(ap1 + 8);
#pragma unroll
    for (int j = 0; j < 4; ++j) {
      const int nt = wave * 4 + j;
      const __bf16* bp = Wp + (((size_t)(nt * 16 + kt)) * 32 + lane) * 16;
      ABfrag B;
      B.h[0] = *(const v8bf*)(bp);
      B.h[1] = *(const v8bf*)(bp + 8);
      C[0][j] = __builtin_amdgcn_wmma_f32_16x16x32_bf16(
          false, A0.v, false, B.v, (short)0, C[0][j], false, false);
      C[1][j] = __builtin_amdgcn_wmma_f32_16x16x32_bf16(
          false, A1.v, false, B.v, (short)0, C[1][j], false, false);
    }
  }

  // ---------- Phase 3: bias (+ReLU) and store ----------
  const int rowoff = (lane >> 4) << 3;     // C layout: vgpr r -> M = r (+8 for hi lanes)
#pragma unroll
  for (int m = 0; m < 2; ++m) {
#pragma unroll
    for (int j = 0; j < 4; ++j) {
      const int   col = (wave * 4 + j) * 16 + (lane & 15);
      const float bv  = bias[col];
#pragma unroll
      for (int r = 0; r < 8; ++r) {
        float v = C[m][j][r] + bv;
        if (RELU) v = v > 0.f ? v : 0.f;
        out[(size_t)(g0 + m * 16 + rowoff + r) * DIM + col] = (TOut)v;
      }
    }
  }
}

extern "C" void kernel_launch(void* const* d_in, const int* in_sizes, int n_in,
                              void* d_out, int out_size, void* d_ws, size_t ws_size,
                              hipStream_t stream) {
  const float* x   = (const float*)d_in[0];
  const int*   adj = (const int*)d_in[1];
  const float* W0  = (const float*)d_in[2];
  const float* b0  = (const float*)d_in[3];
  const float* W1  = (const float*)d_in[4];
  const float* b1  = (const float*)d_in[5];
  const float* W2  = (const float*)d_in[6];
  const float* b2  = (const float*)d_in[7];

  char* ws = (char*)d_ws;
  const size_t WP_ELEMS = (size_t)(2 * DIM) * DIM;   // 512*256 bf16 per layer
  __bf16* Wp0 = (__bf16*)ws;
  __bf16* Wp1 = Wp0 + WP_ELEMS;
  __bf16* Wp2 = Wp1 + WP_ELEMS;

  const size_t HOFF   = (size_t)1 << 20;             // packed weights fit in 1 MB
  const size_t HBYTES = (size_t)NNODES * DIM * sizeof(__bf16);
  __bf16* h1 = (__bf16*)(ws + HOFF);
  __bf16* h2 = (__bf16*)(ws + HOFF + HBYTES);

  // pack per-layer weights into WMMA B-fragment order (bf16)
  dim3 pb(256), pg((unsigned)(WP_ELEMS / 256));
  pack_w_kernel<<<pg, pb, 0, stream>>>(W0, Wp0);
  pack_w_kernel<<<pg, pb, 0, stream>>>(W1, Wp1);
  pack_w_kernel<<<pg, pb, 0, stream>>>(W2, Wp2);

  // 3 SAGE layers; intermediates bf16 (L2-resident), final layer f32 -> d_out
  dim3 lb(128), lg(NNODES / MT);
  sage_layer_kernel<float,  __bf16, true ><<<lg, lb, 0, stream>>>(x,  adj, Wp0, b0, h1);
  sage_layer_kernel<__bf16, __bf16, true ><<<lg, lb, 0, stream>>>(h1, adj, Wp1, b1, h2);
  sage_layer_kernel<__bf16, float,  false><<<lg, lb, 0, stream>>>(h2, adj, Wp2, b2, (float*)d_out);
}
